// MambaHM_20452634263585
// MI455X (gfx1250) — matrix operand order repaired and checked by reading, not compiled, not run on hardware
//
#include <hip/hip_runtime.h>
#include <hip/hip_bf16.h>
#include <math.h>

typedef _Float16 half16 __attribute__((ext_vector_type(16)));
typedef _Float16 half8  __attribute__((ext_vector_type(8)));
typedef float    f32x8  __attribute__((ext_vector_type(8)));

namespace {

constexpr int BZ  = 2;
constexpr int N0  = 16384;
constexpr int NP  = 1024;   // pooled length
constexpr int CHN = 768;
constexpr int C2  = 1536;   // 2*CH
constexpr int DIN = 1536;   // D_INNER
constexpr int DRK = 48;     // DT_RANK
constexpr int NH  = 896;    // TARGET_LEN
constexpr int MM  = BZ * NP;  // 2048 rows for block GEMMs
constexpr int MHH = BZ * NH;  // 1792 rows for head GEMMs

__device__ __forceinline__ float sigm_(float x){ return 1.f/(1.f+__expf(-x)); }
__device__ __forceinline__ float silu_(float x){ return x*sigm_(x); }
__device__ __forceinline__ float gelu_(float x){ return x*sigm_(1.702f*x); }
__device__ __forceinline__ float softplus_(float x){ return (x>20.f)? x : log1pf(__expf(x)); }

// ---------------- stem: conv15 (pad 7) + maxpool16, dna(4ch)->[0,768), seq(1ch)->[768,1536)
__global__ void stem_kernel(const float* __restrict__ dna, const float* __restrict__ seq,
                            const float* __restrict__ wd,  const float* __restrict__ bd,
                            const float* __restrict__ wsq, const float* __restrict__ bs,
                            float* __restrict__ out) // (BZ,NP,C2)
{
  int idx = blockIdx.x*blockDim.x + threadIdx.x;
  if (idx >= BZ*NP*C2) return;
  int c = idx % C2;
  int p = (idx / C2) % NP;
  int b = idx / (C2*NP);
  float best = -3.4e38f;
  if (c < CHN) {
    const float* wrow = wd + c*60; // (CH,4,15)
    for (int j=0;j<16;j++){
      int t = p*16 + j;
      float acc = bd[c];
      for (int k=0;k<15;k++){
        int tt = t + k - 7;
        if (tt >= 0 && tt < N0) {
          const float* xp = dna + ((size_t)(b*N0 + tt))*4;
          acc += xp[0]*wrow[k] + xp[1]*wrow[15+k] + xp[2]*wrow[30+k] + xp[3]*wrow[45+k];
        }
      }
      best = fmaxf(best, acc);
    }
  } else {
    int cc = c - CHN;
    const float* wrow = wsq + cc*15; // (CH,1,15)
    for (int j=0;j<16;j++){
      int t = p*16 + j;
      float acc = bs[cc];
      for (int k=0;k<15;k++){
        int tt = t + k - 7;
        if (tt >= 0 && tt < N0) acc += seq[(size_t)b*N0 + tt]*wrow[k];
      }
      best = fmaxf(best, acc);
    }
  }
  out[idx] = best;
}

// ---------------- LDS fragment assembly (two 16B ds_load_b128 per fragment)
__device__ __forceinline__ half16 frag_lds(const _Float16* rowp, int off0, int off1)
{
  half8 lo = *(const half8*)(rowp + off0);
  half8 hh = *(const half8*)(rowp + off1);
  half16 f;
#pragma unroll
  for (int j=0;j<8;j++){ f[j]=lo[j]; f[j+8]=hh[j]; }
  return f;
}

// ---------------- WMMA GEMM: Y(MxN) = X(MxK, row stride lda) * W(NxK)^T + bias, optional gelu
// 128 threads (4 waves) per 64x64 tile; A/B k-slices staged in LDS as f16;
// each wave computes a 32x32 sub-block = 2x2 v_wmma_f32_16x16x32_f16 per k-step.
__global__ __launch_bounds__(128)
void gemm_wmma_kernel(const float* __restrict__ X, int lda,
                      const float* __restrict__ W,
                      const float* __restrict__ bias,
                      float* __restrict__ Y,
                      int M, int N, int K, int act)
{
  __shared__ _Float16 As[64*32];   // [row][k]   4KB
  __shared__ _Float16 Bs[64*32];   // [col n][k] 4KB
  int tid  = threadIdx.x;
  int lane = tid & 31;
  int wave = tid >> 5;             // 0..3
  int wm = wave >> 1, wn = wave & 1;
  int row = lane & 15, hi = lane >> 4;
  int tmBase = blockIdx.y * 64;
  int tnBase = blockIdx.x * 64;

  // staging: each thread moves 16 f32 -> 16 f16 for A and for B
  int srow  = tid >> 1;            // 0..63
  int skoff = (tid & 1) * 16;      // 0 or 16

  f32x8 acc00 = {}, acc01 = {}, acc10 = {}, acc11 = {};

  for (int k0 = 0; k0 < K; k0 += 32) {
    __syncthreads();
    // ---- stage A slice ----
    {
      int gr = tmBase + srow;
      const float* src = X + (size_t)gr*lda + k0 + skoff;
      half8 h0, h1;
      if (gr < M && k0 + 32 <= K) {
#pragma unroll
        for (int j=0;j<8;j++){ h0[j] = (_Float16)src[j]; h1[j] = (_Float16)src[8+j]; }
      } else {
#pragma unroll
        for (int j=0;j<8;j++){
          int k1 = k0 + skoff + j, k2 = k1 + 8;
          h0[j] = (_Float16)((gr < M && k1 < K) ? src[j]   : 0.f);
          h1[j] = (_Float16)((gr < M && k2 < K) ? src[8+j] : 0.f);
        }
      }
      *(half8*)&As[srow*32 + skoff]     = h0;
      *(half8*)&As[srow*32 + skoff + 8] = h1;
    }
    // ---- stage B slice ----
    {
      int gr = tnBase + srow;
      const float* src = W + (size_t)gr*K + k0 + skoff;
      half8 h0, h1;
      if (gr < N && k0 + 32 <= K) {
#pragma unroll
        for (int j=0;j<8;j++){ h0[j] = (_Float16)src[j]; h1[j] = (_Float16)src[8+j]; }
      } else {
#pragma unroll
        for (int j=0;j<8;j++){
          int k1 = k0 + skoff + j, k2 = k1 + 8;
          h0[j] = (_Float16)((gr < N && k1 < K) ? src[j]   : 0.f);
          h1[j] = (_Float16)((gr < N && k2 < K) ? src[8+j] : 0.f);
        }
      }
      *(half8*)&Bs[srow*32 + skoff]     = h0;
      *(half8*)&Bs[srow*32 + skoff + 8] = h1;
    }
    // ---- prefetch next k slice toward L2 while this one computes ----
    if (k0 + 32 < K) {
      int gr = tmBase + srow;
      if (gr < M) __builtin_prefetch(X + (size_t)gr*lda + k0 + 32 + skoff, 0, 3);
      int gw = tnBase + srow;
      if (gw < N) __builtin_prefetch(W + (size_t)gw*K + k0 + 32 + skoff, 0, 3);
    }
    __syncthreads();

    // ---- fragments from LDS (ISA operand layouts) ----
    // A: halves 0..7 at k=hi*8, halves 8..15 at k=16+hi*8
    half16 a0 = frag_lds(&As[(wm*32 +      row)*32], hi*8, 16 + hi*8);
    half16 a1 = frag_lds(&As[(wm*32 + 16 + row)*32], hi*8, 16 + hi*8);
    // B: halves 0..7 at k=hi*8, halves 8..15 at k=16+hi*8
    half16 b0 = frag_lds(&Bs[(wn*32 +      row)*32], hi*8, 16 + hi*8);
    half16 b1 = frag_lds(&Bs[(wn*32 + 16 + row)*32], hi*8, 16 + hi*8);

    acc00 = __builtin_amdgcn_wmma_f32_16x16x32_f16(false, a0, false, b0, (short)0, acc00, false, false);
    acc01 = __builtin_amdgcn_wmma_f32_16x16x32_f16(false, a0, false, b1, (short)0, acc01, false, false);
    acc10 = __builtin_amdgcn_wmma_f32_16x16x32_f16(false, a1, false, b0, (short)0, acc10, false, false);
    acc11 = __builtin_amdgcn_wmma_f32_16x16x32_f16(false, a1, false, b1, (short)0, acc11, false, false);
  }

  // ---- epilogue: D layout lane = n + 16*(m/8), vgpr = m%8 ----
  {
    int n  = tnBase + wn*32 + row;
    int m0 = tmBase + wm*32 + hi*8;
    if (n < N) {
      float bv = bias ? bias[n] : 0.f;
#pragma unroll
      for (int i=0;i<8;i++){ int m=m0+i; if (m<M){ float v=acc00[i]+bv; if(act==1) v=gelu_(v); Y[(size_t)m*N+n]=v; } }
    }
  }
  {
    int n  = tnBase + wn*32 + 16 + row;
    int m0 = tmBase + wm*32 + hi*8;
    if (n < N) {
      float bv = bias ? bias[n] : 0.f;
#pragma unroll
      for (int i=0;i<8;i++){ int m=m0+i; if (m<M){ float v=acc01[i]+bv; if(act==1) v=gelu_(v); Y[(size_t)m*N+n]=v; } }
    }
  }
  {
    int n  = tnBase + wn*32 + row;
    int m0 = tmBase + wm*32 + 16 + hi*8;
    if (n < N) {
      float bv = bias ? bias[n] : 0.f;
#pragma unroll
      for (int i=0;i<8;i++){ int m=m0+i; if (m<M){ float v=acc10[i]+bv; if(act==1) v=gelu_(v); Y[(size_t)m*N+n]=v; } }
    }
  }
  {
    int n  = tnBase + wn*32 + 16 + row;
    int m0 = tmBase + wm*32 + 16 + hi*8;
    if (n < N) {
      float bv = bias ? bias[n] : 0.f;
#pragma unroll
      for (int i=0;i<8;i++){ int m=m0+i; if (m<M){ float v=acc11[i]+bv; if(act==1) v=gelu_(v); Y[(size_t)m*N+n]=v; } }
    }
  }
}

// ---------------- causal depthwise conv (D_CONV=4) + SiLU, reading channel d of xz(...,3072)
__global__ void dwconv_silu_kernel(const float* __restrict__ xz,
                                   const float* __restrict__ cw, const float* __restrict__ cb,
                                   float* __restrict__ xc) // (BZ,NP,DIN)
{
  int idx = blockIdx.x*blockDim.x + threadIdx.x;
  if (idx >= BZ*NP*DIN) return;
  int d = idx % DIN;
  int t = (idx / DIN) % NP;
  int b = idx / (DIN*NP);
  float acc = cb[d];
#pragma unroll
  for (int k=0;k<4;k++){
    int tt = t - 3 + k;
    if (tt >= 0) acc += xz[((size_t)(b*NP+tt))*(2*DIN) + d] * cw[d*4+k];
  }
  xc[idx] = silu_(acc);
}

// ---------------- selective scan: thread = one (b,d) channel, 16-state recurrence in registers
__global__ void scan_kernel(const float* __restrict__ dtraw, // (BZ,NP,DIN) pre-softplus (bias added)
                            const float* __restrict__ xc,    // (BZ,NP,DIN)
                            const float* __restrict__ xdbl,  // (BZ,NP,80): [48,64)=B, [64,80)=C
                            const float* __restrict__ Alog,  // (DIN,16)
                            const float* __restrict__ Dvec,  // (DIN)
                            float* __restrict__ yb)          // (BZ,NP,DIN)
{
  int chunks = DIN/256;
  int b = blockIdx.x / chunks;
  int d = (blockIdx.x % chunks)*256 + threadIdx.x;
  __shared__ float sB[16];
  __shared__ float sC[16];
  float h[16], A[16];
#pragma unroll
  for (int n=0;n<16;n++){ h[n] = 0.f; A[n] = -__expf(Alog[(size_t)d*16+n]); }
  float Dd = Dvec[d];
  for (int t=0;t<NP;t++){
    __syncthreads();
    if (threadIdx.x < 32) {
      float v = xdbl[((size_t)(b*NP+t))*80 + 48 + threadIdx.x];
      if (threadIdx.x < 16) sB[threadIdx.x] = v; else sC[threadIdx.x-16] = v;
    }
    __syncthreads();
    size_t base = (size_t)(b*NP+t)*DIN + d;
    float dt = softplus_(dtraw[base]);
    float u  = xc[base];
    float du = dt*u;
    float y = 0.f;
#pragma unroll
    for (int n=0;n<16;n++){
      h[n] = __expf(dt*A[n])*h[n] + du*sB[n];
      y += h[n]*sC[n];
    }
    yb[base] = y + u*Dd;
  }
}

// ---------------- y *= silu(z),  z = xz[..., 1536+d]
__global__ void gate_kernel(float* __restrict__ yb, const float* __restrict__ xz)
{
  int idx = blockIdx.x*blockDim.x + threadIdx.x;
  if (idx >= BZ*NP*DIN) return;
  int d = idx % DIN;
  size_t rowb = (size_t)(idx / DIN);
  float z = xz[rowb*(2*DIN) + DIN + d];
  yb[idx] *= silu_(z);
}

// ---------------- residual accumulate + dual RMS norm (fwd half / channel-flipped rev half)
__global__ void resid_rms_kernel(const float* __restrict__ x, float* __restrict__ R,
                                 const float* __restrict__ nw, float* __restrict__ xn, int first)
{
  int bt  = blockIdx.x;       // 0..MM-1
  int tid = threadIdx.x;      // 256 threads, 6 channels each
  __shared__ float sh[256];
  size_t base = (size_t)bt * C2;
  float v[6]; float s = 0.f;
#pragma unroll
  for (int i=0;i<6;i++){
    int c = tid*6 + i;
    float r = x[base+c];
    if (!first) r += R[base+c];
    R[base+c] = r;
    v[i] = r;
    s += r*r;
  }
  sh[tid] = s;
  __syncthreads();
  for (int off=64; off>=1; off>>=1){          // reduce halves [0,128) / [128,256)
    if ((tid & 127) < off) sh[tid] += sh[tid+off];
    __syncthreads();
  }
  float inv_f = rsqrtf(sh[0]  /(float)CHN + 1e-5f);
  float inv_r = rsqrtf(sh[128]/(float)CHN + 1e-5f);
#pragma unroll
  for (int i=0;i<6;i++){
    int c = tid*6 + i;
    xn[base+c] = (c < CHN) ? v[i]*inv_f*nw[c] : v[i]*inv_r*nw[C2-1-c];
  }
}

// ---------------- generic strided/flipped copy: dst[b,t,dst_off+c] = src[b,t',src_off+c']
__global__ void copy_perm_kernel(float* __restrict__ dst, int dstC, int dst_off,
                                 const float* __restrict__ src, int srcC, int src_off,
                                 int T, int C, int flip_t, int flip_c)
{
  int idx = blockIdx.x*blockDim.x + threadIdx.x;
  if (idx >= BZ*T*C) return;
  int c = idx % C;
  int t = (idx / C) % T;
  int b = idx / (C*T);
  int ts = flip_t ? (T-1-t) : t;
  int cs = flip_c ? (C-1-c) : c;
  dst[((size_t)(b*T+t))*dstC + dst_off + c] = src[((size_t)(b*T+ts))*srcC + src_off + cs];
}

// ---------------- out[b,t,c] = A[b,t,c] + Bm[b,T-1-t,c]
__global__ void add_tflip_kernel(const float* __restrict__ Am, const float* __restrict__ Bm,
                                 float* __restrict__ out, int T, int C)
{
  int idx = blockIdx.x*blockDim.x + threadIdx.x;
  if (idx >= BZ*T*C) return;
  int c = idx % C;
  int t = (idx / C) % T;
  int b = idx / (C*T);
  out[idx] = Am[idx] + Bm[((size_t)(b*T + (T-1-t)))*C + c];
}

// ---------------- heads: trim 64, batchnorm affine, gelu
__global__ void head_pre_kernel(const float* __restrict__ x,
                                const float* __restrict__ g, const float* __restrict__ bb,
                                const float* __restrict__ rm, const float* __restrict__ rv,
                                float* __restrict__ H1) // (BZ,NH,C2)
{
  int idx = blockIdx.x*blockDim.x + threadIdx.x;
  if (idx >= BZ*NH*C2) return;
  int c = idx % C2;
  int t = (idx / C2) % NH;
  int b = idx / (C2*NH);
  float v = x[((size_t)(b*NP + t + 64))*C2 + c];
  v = (v - rm[c]) * rsqrtf(rv[c] + 1e-3f) * g[c] + bb[c];
  H1[idx] = gelu_(v);
}

// ---------------- final 1536->1 projection + softplus
__global__ void head_out_kernel(const float* __restrict__ H2, const float* __restrict__ ow,
                                const float* __restrict__ ob, float* __restrict__ out)
{
  int bt = blockIdx.x;   // 0..MHH-1
  int tid = threadIdx.x; // 256
  __shared__ float sh[256];
  float s = 0.f;
  for (int c = tid; c < C2; c += 256) s += H2[(size_t)bt*C2 + c] * ow[c];
  sh[tid] = s;
  __syncthreads();
  for (int off=128; off>=1; off>>=1){
    if (tid < off) sh[tid] += sh[tid+off];
    __syncthreads();
  }
  if (tid == 0) out[bt] = softplus_(sh[0] + ob[0]);
}

// =======================================================================
struct LayerW {
  const float *norm_w, *in_proj, *conv_w, *conv_b, *x_proj, *dt_w, *dt_b, *A_log, *D, *out_proj;
};

struct Bufs {
  float *S, *X, *R, *XN, *HBUF, *HFL, *M1, *M2, *FOUT, *GOUT;
  float *XZ, *XC, *XDBL, *DTB, *YB, *H1, *H2;
};

inline void launch_gemm(hipStream_t st, const float* Xp, int lda, const float* Wp,
                        const float* bias, float* Yp, int Mv, int Nv, int Kv, int act)
{
  dim3 grid((Nv + 63)/64, (Mv + 63)/64);
  gemm_wmma_kernel<<<grid, dim3(128), 0, st>>>(Xp, lda, Wp, bias, Yp, Mv, Nv, Kv, act);
}

inline void run_mamba(hipStream_t st, const float* H, const LayerW& L, const Bufs& B, float* out)
{
  launch_gemm(st, H, CHN, L.in_proj, nullptr, B.XZ, MM, 2*DIN, CHN, 0);
  dwconv_silu_kernel<<<(MM*DIN)/256, 256, 0, st>>>(B.XZ, L.conv_w, L.conv_b, B.XC);
  launch_gemm(st, B.XC, DIN, L.x_proj, nullptr, B.XDBL, MM, 80, DIN, 0);
  launch_gemm(st, B.XDBL, 80, L.dt_w, L.dt_b, B.DTB, MM, DIN, DRK, 0);
  scan_kernel<<<BZ*(DIN/256), 256, 0, st>>>(B.DTB, B.XC, B.XDBL, L.A_log, L.D, B.YB);
  gate_kernel<<<(MM*DIN)/256, 256, 0, st>>>(B.YB, B.XZ);
  launch_gemm(st, B.YB, DIN, L.out_proj, nullptr, out, MM, CHN, DIN, 0);
}

inline void run_bi_mamba(hipStream_t st, const float* H, const LayerW& L, const Bufs& B, float* out)
{
  const int nblk = (BZ*NP*CHN)/256;
  run_mamba(st, H, L, B, B.M1);
  copy_perm_kernel<<<nblk, 256, 0, st>>>(B.HFL, CHN, 0, H, CHN, 0, NP, CHN, 1, 0);
  run_mamba(st, B.HFL, L, B, B.M2);
  add_tflip_kernel<<<nblk, 256, 0, st>>>(B.M1, B.M2, out, NP, CHN);
}

inline void run_block(hipStream_t st, const LayerW& L, const Bufs& B, int first)
{
  const int nblk = (BZ*NP*CHN)/256;
  resid_rms_kernel<<<MM, 256, 0, st>>>(B.X, B.R, L.norm_w, B.XN, first);
  // forward half
  copy_perm_kernel<<<nblk, 256, 0, st>>>(B.HBUF, CHN, 0, B.XN, C2, 0, NP, CHN, 0, 0);
  run_bi_mamba(st, B.HBUF, L, B, B.FOUT);
  // reverse half: flip(time, channel) of xn[..., CH:]
  copy_perm_kernel<<<nblk, 256, 0, st>>>(B.HBUF, CHN, 0, B.XN, C2, CHN, NP, CHN, 1, 1);
  run_bi_mamba(st, B.HBUF, L, B, B.GOUT);
  // x = concat(out_f, flip(out_r))
  copy_perm_kernel<<<nblk, 256, 0, st>>>(B.X, C2, 0,   B.FOUT, CHN, 0, NP, CHN, 0, 0);
  copy_perm_kernel<<<nblk, 256, 0, st>>>(B.X, C2, CHN, B.GOUT, CHN, 0, NP, CHN, 1, 1);
}

} // namespace

extern "C" void kernel_launch(void* const* d_in, const int* in_sizes, int n_in,
                              void* d_out, int out_size, void* d_ws, size_t ws_size,
                              hipStream_t stream)
{
  if (n_in < 36) return;
  // setup_inputs() dict insertion order, params flattened recursively
  const float* dna        = (const float*)d_in[0];
  const float* seq        = (const float*)d_in[1];
  const float* stem_w     = (const float*)d_in[2];
  const float* stem_b     = (const float*)d_in[3];
  const float* stem_seq_w = (const float*)d_in[4];
  const float* stem_seq_b = (const float*)d_in[5];
  const float* fuse_w     = (const float*)d_in[6];
  const float* fuse_b     = (const float*)d_in[7];
  LayerW L[2];
  for (int l = 0; l < 2; l++) {
    int base = 8 + l*10;
    L[l].norm_w   = (const float*)d_in[base+0];
    L[l].in_proj  = (const float*)d_in[base+1];
    L[l].conv_w   = (const float*)d_in[base+2];
    L[l].conv_b   = (const float*)d_in[base+3];
    L[l].x_proj   = (const float*)d_in[base+4];
    L[l].dt_w     = (const float*)d_in[base+5];
    L[l].dt_b     = (const float*)d_in[base+6];
    L[l].A_log    = (const float*)d_in[base+7];
    L[l].D        = (const float*)d_in[base+8];
    L[l].out_proj = (const float*)d_in[base+9];
  }
  const float* bn_g        = (const float*)d_in[28];
  const float* bn_b        = (const float*)d_in[29];
  const float* bn_rm       = (const float*)d_in[30];
  const float* bn_rv       = (const float*)d_in[31];
  const float* head_conv_w = (const float*)d_in[32];
  const float* head_conv_b = (const float*)d_in[33];
  const float* out_w       = (const float*)d_in[34];
  const float* out_b       = (const float*)d_in[35];
  float* outp = (float*)d_out;

  // ---- workspace layout (floats) ----
  float* ws = (float*)d_ws;
  const size_t szMC2 = (size_t)MM*C2;      // 3,145,728
  const size_t szMCH = (size_t)MM*CHN;     // 1,572,864
  Bufs B;
  size_t o = 0;
  B.S    = ws + o; o += szMC2;             // stem/fuse input; reused as H1
  B.X    = ws + o; o += szMC2;
  B.R    = ws + o; o += szMC2;
  B.XN   = ws + o; o += szMC2;
  B.HBUF = ws + o; o += szMCH;
  B.HFL  = ws + o; o += szMCH;
  B.M1   = ws + o; o += szMCH;
  B.M2   = ws + o; o += szMCH;
  B.FOUT = ws + o; o += szMCH;
  B.GOUT = ws + o; o += szMCH;
  B.XZ   = ws + o; o += (size_t)MM*2*DIN;  // reused as H2
  B.XC   = ws + o; o += (size_t)MM*DIN;
  B.XDBL = ws + o; o += (size_t)MM*80;
  B.DTB  = ws + o; o += (size_t)MM*DIN;
  B.YB   = ws + o; o += (size_t)MM*DIN;
  if (ws_size < o * sizeof(float)) return; // ~152 MB required
  B.H1 = B.S;   // (MHH,C2) fits in S region
  B.H2 = B.XZ;  // (MHH,C2) fits in XZ region

  // ---- stem + fuse ----
  stem_kernel<<<(BZ*NP*C2)/256, 256, 0, stream>>>(dna, seq, stem_w, stem_b,
                                                  stem_seq_w, stem_seq_b, B.S);
  launch_gemm(stream, B.S, C2, fuse_w, fuse_b, B.X, MM, C2, C2, 0);

  // ---- mamba blocks ----
  run_block(stream, L[0], B, 1);
  run_block(stream, L[1], B, 0);

  // ---- heads ----
  head_pre_kernel<<<(BZ*NH*C2)/256, 256, 0, stream>>>(B.X, bn_g, bn_b, bn_rm, bn_rv, B.H1);
  launch_gemm(stream, B.H1, C2, head_conv_w, head_conv_b, B.H2, MHH, C2, C2, 1 /*gelu*/);
  head_out_kernel<<<MHH, 256, 0, stream>>>(B.H2, out_w, out_b, outp);
  (void)in_sizes; (void)out_size;
}
